// MultiHeadRouter_52544629899284
// MI455X (gfx1250) — compile-verified, matrix operand order checked by reading, not run on hardware
//
#include <hip/hip_runtime.h>
#include <stdint.h>

// ---------------------------------------------------------------------------
// MultiHeadRouter for MI455X (gfx1250), wave32 + V_WMMA_F32_16X16X32_BF16.
// Round 2: 64-token M-tile, 2x4 accumulator block per wave (B-fragment reuse),
// double-buffered LDS staging so global loads overlap the WMMA section.
//
// Sizes: x [16384 tokens, 4096], W [256 rows (g*64+e), 4096].
// Out layout (flat, return order):
//   idx    [16384,2]  @ 0        (stored as float, harness float* output)
//   scores [16384,2]  @ 32768
//   probs  [16384,64] @ 65536
//   imp    [64]       @ 1114112
//   load   [64]       @ 1114176
// ---------------------------------------------------------------------------

typedef __attribute__((ext_vector_type(16))) __bf16 v16bf;
typedef __attribute__((ext_vector_type(8)))  float  v8f;

#define DDIM     4096
#define NEXP     256        // G*E columns
#define MT       64         // tokens per workgroup
#define KC       64         // K chunk staged through LDS
#define NCH      (DDIM / KC)
#define OFF_SCORES 32768
#define OFF_PROBS  65536
#define OFF_IMP    1114112
#define OFF_LOAD   1114176
#define INV_TOK  (1.0f / 16384.0f)

// LDS layout (dynamic, 147456 B):
//   [0      ..  65536)  float lg[64][256]        (logits after GEMM)
//   [65536  ..  81920)  bf16  A [2][64][64]      (ping-pong; reused as probs)
//   [81920  .. 147456)  bf16  B [2][256][64]     (ping-pong)
#define LDS_A_OFF  65536
#define LDS_B_OFF  81920
#define LDS_BYTES  147456

__device__ __forceinline__ __bf16 f2bf(float f) {
  union { float f; uint32_t u; } v; v.f = f;
  uint32_t r = v.u + 0x7FFFu + ((v.u >> 16) & 1u);   // round-to-nearest-even
  union { uint16_t u; __bf16 b; } o; o.u = (uint16_t)(r >> 16);
  return o.b;
}

// Kernel 0: convert W (fp32, [256][4096]) -> bf16 in d_ws; zero imp/load stats.
__global__ void router_convW_kernel(const float* __restrict__ W,
                                    uint16_t* __restrict__ Wb,
                                    float* __restrict__ out) {
  if (blockIdx.x == 0 && threadIdx.x < 128) out[OFF_IMP + threadIdx.x] = 0.0f;
  int i = (blockIdx.x * 256 + threadIdx.x) * 4;
  #pragma unroll
  for (int q = 0; q < 4; ++q) {
    union { __bf16 b; uint16_t u; } c; c.b = f2bf(W[i + q]);
    Wb[i + q] = c.u;
  }
}

// Kernel 1: fused GEMM (bf16 WMMA) + softmax/mean/top-2/stats epilogue.
__launch_bounds__(256, 1)
__global__ void router_main_kernel(const float* __restrict__ x,
                                   const uint16_t* __restrict__ Wb16,
                                   float* __restrict__ out) {
  extern __shared__ char smemRaw[];
  float*  lgS = (float*)smemRaw;                  // [64][256]
  __bf16* aS  = (__bf16*)(smemRaw + LDS_A_OFF);   // [2][64][64]
  __bf16* bS  = (__bf16*)(smemRaw + LDS_B_OFF);   // [2][256][64]
  float*  pS  = (float*)(smemRaw + LDS_A_OFF);    // [64][64] (reuse A buffers)

  const __bf16* Wb = (const __bf16*)Wb16;
  const int tid  = threadIdx.x;
  const int lane = tid & 31;
  const int wave = tid >> 5;
  const int tok0 = blockIdx.x * MT;

  // 8 waves: 2 row-pairs x 4 col-quads; each wave owns 2x4 16x16 tiles.
  const int rowP  = wave & 1;            // rows {2*rowP, 2*rowP+1} of 4 row tiles
  const int colT0 = (wave >> 1) * 4;     // 4 consecutive 16-wide col tiles
  const int hlf   = lane >> 4;           // bf16 fragment half (K-group select)
  const int l16   = lane & 15;

  v8f acc[2][4];
  #pragma unroll
  for (int r = 0; r < 2; ++r)
    #pragma unroll
    for (int c = 0; c < 4; ++c)
      #pragma unroll
      for (int i = 0; i < 8; ++i) acc[r][c][i] = 0.0f;

  // cooperative staging: 256 threads cover A(64x64 fp32) and B(256x64 bf16)
  const int aRowLd = tid >> 2;           // 0..63
  const int aKLd   = (tid & 3) * 16;     // 0,16,32,48
  float4 ax[4];                          // 16 fp32 of x per thread per chunk
  uint4  bx[8];                          // 64 bf16 of W per thread per chunk

  auto loadRegs = [&](int c) {
    const float* xp = x + (size_t)(tok0 + aRowLd) * DDIM + c * KC + aKLd;
    #pragma unroll
    for (int q = 0; q < 4; ++q) ax[q] = *(const float4*)(xp + q * 4);
    const uint4* wp = (const uint4*)(Wb + (size_t)tid * DDIM + c * KC);
    #pragma unroll
    for (int q = 0; q < 8; ++q) bx[q] = wp[q];
  };

  auto storeLDS = [&](int buf) {
    __bf16* a = aS + buf * (MT * KC);
    #pragma unroll
    for (int q = 0; q < 2; ++q) {
      union { __bf16 b[8]; uint4 u; } pk;
      const float* f = (const float*)&ax[q * 2];
      #pragma unroll
      for (int j = 0; j < 8; ++j) pk.b[j] = f2bf(f[j]);
      *(uint4*)(a + aRowLd * KC + aKLd + q * 8) = pk.u;
    }
    uint4* dst = (uint4*)(bS + buf * (NEXP * KC) + tid * KC);
    #pragma unroll
    for (int q = 0; q < 8; ++q) dst[q] = bx[q];
  };

  auto compute = [&](int buf) {
    const __bf16* a = aS + buf * (MT * KC);
    const __bf16* b = bS + buf * (NEXP * KC);
    #pragma unroll
    for (int ks = 0; ks < 2; ++ks) {
      const int klo = ks * 32 + hlf * 8;      // K {0-7,16-23} / {8-15,24-31}
      v16bf af[2];
      #pragma unroll
      for (int r = 0; r < 2; ++r) {
        const int row = (rowP * 2 + r) * 16 + l16;
        uint4* ap = (uint4*)&af[r];
        ap[0] = *(const uint4*)(a + row * KC + klo);
        ap[1] = *(const uint4*)(a + row * KC + klo + 16);
      }
      #pragma unroll
      for (int cc = 0; cc < 4; ++cc) {
        const int col = (colT0 + cc) * 16 + l16;   // lane = B column
        v16bf bfr;
        uint4* bp = (uint4*)&bfr;
        bp[0] = *(const uint4*)(b + col * KC + klo);
        bp[1] = *(const uint4*)(b + col * KC + klo + 16);
        #pragma unroll
        for (int r = 0; r < 2; ++r)              // B fragment reused twice
          acc[r][cc] = __builtin_amdgcn_wmma_f32_16x16x32_bf16(
              false, af[r], false, bfr, (short)0, acc[r][cc], false, false);
      }
    }
  };

  // ---- software-pipelined GEMM: global loads for c+1 overlap WMMAs on c ----
  loadRegs(0);
  storeLDS(0);
  __syncthreads();
  for (int c = 0; c < NCH; ++c) {
    const int cur = c & 1;
    const bool more = (c + 1 < NCH);
    if (more) loadRegs(c + 1);   // vmem issues here, waited on in storeLDS
    compute(cur);                // 16 WMMAs of latency cover
    if (more) storeLDS(cur ^ 1);
    __syncthreads();
  }

  // ---- dump accumulators: VGPR i -> token row (i + 8*hlf), col = l16 ----
  #pragma unroll
  for (int r = 0; r < 2; ++r)
    #pragma unroll
    for (int c = 0; c < 4; ++c) {
      const int col = (colT0 + c) * 16 + l16;
      #pragma unroll
      for (int i = 0; i < 8; ++i) {
        const int trow = (rowP * 2 + r) * 16 + i + 8 * hlf;
        lgS[trow * NEXP + col] = acc[r][c][i];
      }
    }
  __syncthreads();

  // ---- epilogue: one token per thread (threads 0..63) ----
  if (tid < MT) {
    float* pf = pS + tid * 64;
    for (int e = 0; e < 64; ++e) pf[e] = 0.0f;
    for (int g = 0; g < 4; ++g) {
      float* lg = lgS + tid * NEXP + g * 64;
      float mx = -3.4e38f;
      for (int e = 0; e < 64; ++e) mx = fmaxf(mx, lg[e]);
      float s = 0.0f;
      for (int e = 0; e < 64; ++e) { float v = __expf(lg[e] - mx); lg[e] = v; s += v; }
      const float inv = 0.25f / s;               // head-mean folded in
      for (int e = 0; e < 64; ++e) pf[e] += lg[e] * inv;
    }
    const int tok = tok0 + tid;
    float* op = out + OFF_PROBS + (size_t)tok * 64;
    for (int e = 0; e < 64; ++e) op[e] = pf[e];
    // top-2 (first occurrence wins ties, descending, like lax.top_k)
    float m1 = -1.0f, m2 = -1.0f; int i1 = 0, i2 = 0;
    for (int e = 0; e < 64; ++e) {
      float v = pf[e];
      if (v > m1)      { m2 = m1; i2 = i1; m1 = v; i1 = e; }
      else if (v > m2) { m2 = v; i2 = e; }
    }
    const float den = fmaxf(m1 + m2, 1e-9f);
    out[2 * tok + 0] = (float)i1;
    out[2 * tok + 1] = (float)i2;
    out[OFF_SCORES + 2 * tok + 0] = m1 / den;
    out[OFF_SCORES + 2 * tok + 1] = m2 / den;
  }
  __syncthreads();
  // ---- importance / load: per-WG partial sums, one atomic per expert ----
  if (tid < 64) {
    float s = 0.0f, c = 0.0f;
    for (int t = 0; t < MT; ++t) {
      float v = pS[t * 64 + tid];
      s += v;
      c += (v > 0.0f) ? 1.0f : 0.0f;
    }
    atomicAdd(out + OFF_IMP  + tid, s * INV_TOK);
    atomicAdd(out + OFF_LOAD + tid, c * INV_TOK);
  }
}

extern "C" void kernel_launch(void* const* d_in, const int* in_sizes, int n_in,
                              void* d_out, int out_size, void* d_ws, size_t ws_size,
                              hipStream_t stream) {
  const float* x = (const float*)d_in[0];          // [4,4096,4096] fp32
  const float* W = (const float*)d_in[1];          // [4,64,4096]  fp32
  float* out = (float*)d_out;
  uint16_t* Wb = (uint16_t*)d_ws;                  // 2 MB bf16 W in scratch

  // W fp32 -> bf16 (1M elems: 1024 blocks x 256 thr x 4) + zero stats region
  router_convW_kernel<<<dim3(1024), dim3(256), 0, stream>>>(W, Wb, out);
  // 16384 tokens / 64 per WG = 256 WGs
  router_main_kernel<<<dim3(256), dim3(256), LDS_BYTES, stream>>>(x, Wb, out);
}
